// QuantumQKGenerator_59545426592465
// MI455X (gfx1250) — matrix operand order repaired and checked by reading, not compiled, not run on hardware
//
#include <hip/hip_runtime.h>
#include <hip/hip_bf16.h>
#include <math.h>

#define NQ   10
#define DIM  1024
#define XLEN 768

typedef _Float16 v16h __attribute__((ext_vector_type(16)));
typedef _Float16 v8h  __attribute__((ext_vector_type(8)));
typedef float    v8f  __attribute__((ext_vector_type(8)));

// ---------------------------------------------------------------------------
// Kernel 1: normalize + embed, apply RZ*RY*RX per qubit, CNOT ring, RY layer.
// One block per batch row; statevector lives in LDS. Outputs f16 Re/Im planes.
// qubit q <-> bit (9-q) of the flat index (wire 0 is MSB in the reference).
// ---------------------------------------------------------------------------
__global__ __launch_bounds__(256) void k_prep(const float* __restrict__ x,
                                              const float* __restrict__ w,
                                              _Float16* __restrict__ Sr,
                                              _Float16* __restrict__ Si) {
  __shared__ float sr[DIM], si[DIM];
  __shared__ float red[256];
  __shared__ float g[NQ][8];   // combined U = RZ*RY*RX: u00,u01,u10,u11 (re,im)
  __shared__ float g2[NQ][2];  // final RY: cos, sin

  const int row = blockIdx.x;
  const int t   = threadIdx.x;
  const float* xr = x + (size_t)row * XLEN;

  // ---- L2 norm over the 768 real inputs -------------------------------
  float acc = 0.f;
  for (int i = t; i < XLEN; i += 256) { float v = xr[i]; acc += v * v; }
  red[t] = acc;
  __syncthreads();
  for (int s = 128; s > 0; s >>= 1) {
    if (t < s) red[t] += red[t + s];
    __syncthreads();
  }

  // ---- per-qubit gate matrices (threads 0..9) -------------------------
  if (t < NQ) {
    float a = w[t * 4 + 0], b = w[t * 4 + 1], c = w[t * 4 + 2], d = w[t * 4 + 3];
    float ca, sa, cb, sb, cc, sc, cd, sd;
    __sincosf(0.5f * a, &sa, &ca);
    __sincosf(0.5f * b, &sb, &cb);
    __sincosf(0.5f * c, &sc, &cc);
    __sincosf(0.5f * d, &sd, &cd);
    // M = RY(b)*RX(a)
    float t00r =  cb * ca, t00i =  sb * sa;
    float t01r = -sb * ca, t01i = -cb * sa;
    float t10r =  sb * ca, t10i = -cb * sa;
    float t11r =  cb * ca, t11i = -sb * sa;
    // U = RZ(c)*M ; e- = (cc,-sc) on row0, e+ = (cc,+sc) on row1
    g[t][0] = cc * t00r + sc * t00i;  g[t][1] = cc * t00i - sc * t00r;
    g[t][2] = cc * t01r + sc * t01i;  g[t][3] = cc * t01i - sc * t01r;
    g[t][4] = cc * t10r - sc * t10i;  g[t][5] = cc * t10i + sc * t10r;
    g[t][6] = cc * t11r - sc * t11i;  g[t][7] = cc * t11i + sc * t11r;
    g2[t][0] = cd; g2[t][1] = sd;
  }
  __syncthreads();

  const float inv = 1.0f / fmaxf(sqrtf(red[0]), 1e-8f);
  for (int i = t; i < DIM; i += 256) {
    sr[i] = (i < XLEN) ? xr[i] * inv : 0.f;
    si[i] = 0.f;
  }
  __syncthreads();

  // ---- combined RZ*RY*RX per qubit ------------------------------------
  for (int q = 0; q < NQ; ++q) {
    const int pos = 9 - q, bit = 1 << pos;
    const float u00r = g[q][0], u00i = g[q][1], u01r = g[q][2], u01i = g[q][3];
    const float u10r = g[q][4], u10i = g[q][5], u11r = g[q][6], u11i = g[q][7];
    for (int p = t; p < DIM / 2; p += 256) {
      int i0 = ((p >> pos) << (pos + 1)) | (p & (bit - 1));
      int i1 = i0 | bit;
      float a0r = sr[i0], a0i = si[i0], a1r = sr[i1], a1i = si[i1];
      float n0r = u00r * a0r - u00i * a0i + u01r * a1r - u01i * a1i;
      float n0i = u00r * a0i + u00i * a0r + u01r * a1i + u01i * a1r;
      float n1r = u10r * a0r - u10i * a0i + u11r * a1r - u11i * a1i;
      float n1i = u10r * a0i + u10i * a0r + u11r * a1i + u11i * a1r;
      sr[i0] = n0r; si[i0] = n0i; sr[i1] = n1r; si[i1] = n1i;
    }
    __syncthreads();
  }

  // ---- CNOT ring: control q, target (q+1)%10 --------------------------
  for (int q = 0; q < NQ; ++q) {
    const int pc = 9 - q, pt = 9 - ((q + 1) % NQ);
    const int cbit = 1 << pc, tbit = 1 << pt;
    const int b1 = (pc < pt) ? pc : pt, b2 = (pc < pt) ? pt : pc;
    // 256 pairs: 8 free bits; insert zeros at b1 then b2, set control bit
    {
      int p = t;
      int xm = ((p >> b1) << (b1 + 1)) | (p & ((1 << b1) - 1));
      xm = ((xm >> b2) << (b2 + 1)) | (xm & ((1 << b2) - 1));
      xm |= cbit;                       // control=1, target=0
      int xp = xm | tbit;
      float tr = sr[xm], ti = si[xm];
      sr[xm] = sr[xp]; si[xm] = si[xp];
      sr[xp] = tr;     si[xp] = ti;
    }
    __syncthreads();
  }

  // ---- final RY layer --------------------------------------------------
  for (int q = 0; q < NQ; ++q) {
    const int pos = 9 - q, bit = 1 << pos;
    const float c = g2[q][0], s = g2[q][1];
    for (int p = t; p < DIM / 2; p += 256) {
      int i0 = ((p >> pos) << (pos + 1)) | (p & (bit - 1));
      int i1 = i0 | bit;
      float a0r = sr[i0], a0i = si[i0], a1r = sr[i1], a1i = si[i1];
      sr[i0] = c * a0r - s * a1r;  si[i0] = c * a0i - s * a1i;
      sr[i1] = s * a0r + c * a1r;  si[i1] = s * a0i + c * a1i;
    }
    __syncthreads();
  }

  _Float16* dr = Sr + (size_t)row * DIM;
  _Float16* di = Si + (size_t)row * DIM;
  for (int i = t; i < DIM; i += 256) {
    dr[i] = (_Float16)sr[i];
    di[i] = (_Float16)si[i];
  }
}

// ---------------------------------------------------------------------------
// Kernel 2: QFT matrix (unscaled): F[x][y] = exp(2*pi*i*x*y/1024), f16 planes.
// Symmetric, so it serves as both row-major and column-major operand.
// ---------------------------------------------------------------------------
__global__ __launch_bounds__(256) void k_qft(_Float16* __restrict__ Fr,
                                             _Float16* __restrict__ Fi) {
  const int idx = blockIdx.x * 256 + threadIdx.x;
  const int xc = idx >> 10, y = idx & (DIM - 1);
  const int ph = (xc * y) & (DIM - 1);          // phase mod 2*pi for precision
  float s, c;
  __sincosf((float)ph * (6.28318530717958647692f / 1024.0f), &s, &c);
  Fr[idx] = (_Float16)c;
  Fi[idx] = (_Float16)s;
}

// ---------------------------------------------------------------------------
// Kernel 3: complex GEMM via 4 real WMMA f16->f32 GEMMs.
//   Or = (Sr*Fr - Si*Fi)/32 ;  Oi = (Sr*Fi + Si*Fr)/32
// Block = 256 threads (8 waves); block tile 128(M) x 64(N); K-loop over 1024.
// Inner loop issues WMMAs in 4 passes over the N-subtiles so that any two
// WMMAs updating the same accumulator are ~7 WMMAs apart (hides the f16
// WMMA->WMMA hazard window without v_nops).
// ---------------------------------------------------------------------------
__device__ __forceinline__ v16h load_afrag(const _Float16* __restrict__ base,
                                           int m, int kb, int lane) {
  // A 16x32: lanes 0-15 -> K {0..7,16..23}; lanes 16-31 -> K {8..15,24..31}
  const int koff = (lane >> 4) << 3;
  const _Float16* p = base + (size_t)m * DIM + kb + koff;
  v8h lo = *(const v8h*)(p);
  v8h hi = *(const v8h*)(p + 16);
  v16h r;
#pragma unroll
  for (int i = 0; i < 8; ++i) { r[i] = lo[i]; r[i + 8] = hi[i]; }
  return r;
}

__device__ __forceinline__ v16h load_bfrag(const _Float16* __restrict__ F,
                                           int n, int kb, int lane) {
  // B 32x16: lanes 0-15 -> col n, K {0..7,16..23}; lanes 16-31 -> col n, K {8..15,24..31}.
  // F is symmetric: column n == row n.
  const int koff = (lane >> 4) << 3;
  const _Float16* p = F + (size_t)n * DIM + kb + koff;
  v8h lo = *(const v8h*)(p);
  v8h hi = *(const v8h*)(p + 16);
  v16h r;
#pragma unroll
  for (int i = 0; i < 8; ++i) { r[i] = lo[i]; r[i + 8] = hi[i]; }
  return r;
}

__global__ __launch_bounds__(256) void k_gemm(const _Float16* __restrict__ Sr,
                                              const _Float16* __restrict__ Si,
                                              const _Float16* __restrict__ Fr,
                                              const _Float16* __restrict__ Fi,
                                              float* __restrict__ Or,
                                              float* __restrict__ Oi) {
  const int lane = threadIdx.x & 31;
  const int wave = threadIdx.x >> 5;
  const int m0 = blockIdx.x * 128 + wave * 16;
  const int n0 = blockIdx.y * 64;
  const int mrow = m0 + (lane & 15);

  v8f accR[4], accI[4];
#pragma unroll
  for (int j = 0; j < 4; ++j) {
#pragma unroll
    for (int r = 0; r < 8; ++r) { accR[j][r] = 0.f; accI[j][r] = 0.f; }
  }

  for (int kb = 0; kb < DIM; kb += 32) {
    v16h ar = load_afrag(Sr, mrow, kb, lane);
    v16h ai = load_afrag(Si, mrow, kb, lane);
    v16h ani;                                  // -Ai (f16 WMMA has no A-neg)
#pragma unroll
    for (int i = 0; i < 16; ++i) ani[i] = (_Float16)(-(float)ai[i]);

    if (kb + 32 < DIM)                          // hint next F rows into cache
      __builtin_prefetch(Fr + (size_t)(n0 + (lane & 15)) * DIM + kb + 32, 0, 0);

    v16h br[4], bi[4];
#pragma unroll
    for (int j = 0; j < 4; ++j) {
      const int n = n0 + j * 16 + (lane & 15);
      br[j] = load_bfrag(Fr, n, kb, lane);
      bi[j] = load_bfrag(Fi, n, kb, lane);
    }

    // Pass 1: accR += Ar*Br            (independent across j)
#pragma unroll
    for (int j = 0; j < 4; ++j)
      accR[j] = __builtin_amdgcn_wmma_f32_16x16x32_f16(false, ar, false, br[j],
                                                       (short)0, accR[j], false, false);
    // Pass 2: accI += Ar*Bi
#pragma unroll
    for (int j = 0; j < 4; ++j)
      accI[j] = __builtin_amdgcn_wmma_f32_16x16x32_f16(false, ar, false, bi[j],
                                                       (short)0, accI[j], false, false);
    // Pass 3: accR += (-Ai)*Bi   (7 WMMAs since accR[j]'s last update)
#pragma unroll
    for (int j = 0; j < 4; ++j)
      accR[j] = __builtin_amdgcn_wmma_f32_16x16x32_f16(false, ani, false, bi[j],
                                                       (short)0, accR[j], false, false);
    // Pass 4: accI += Ai*Br
#pragma unroll
    for (int j = 0; j < 4; ++j)
      accI[j] = __builtin_amdgcn_wmma_f32_16x16x32_f16(false, ai, false, br[j],
                                                       (short)0, accI[j], false, false);
  }

  // C/D layout: VGPR r -> row m0 + r + 8*(lane>=16), col = n0 + (lane&15)
  const float scale = 0.03125f;                 // 1/sqrt(1024)
  const int rbase = m0 + ((lane >> 4) << 3);
  const int col = (lane & 15);
#pragma unroll
  for (int j = 0; j < 4; ++j) {
#pragma unroll
    for (int r = 0; r < 8; ++r) {
      size_t o = (size_t)(rbase + r) * DIM + n0 + j * 16 + col;
      Or[o] = accR[j][r] * scale;
      Oi[o] = accI[j][r] * scale;
    }
  }
}

// ---------------------------------------------------------------------------
// Kernel 4: expectation values <X_q>,<Y_q>,<Z_q> per row -> (B, 30) f32.
// ---------------------------------------------------------------------------
__global__ __launch_bounds__(256) void k_expect(const float* __restrict__ Or,
                                                const float* __restrict__ Oi,
                                                float* __restrict__ out) {
  __shared__ float sr[DIM], si[DIM];
  __shared__ float rx[256], ry[256], rz[256];
  const int row = blockIdx.x, t = threadIdx.x;
  const float* pr = Or + (size_t)row * DIM;
  const float* pi = Oi + (size_t)row * DIM;
  for (int i = t; i < DIM; i += 256) { sr[i] = pr[i]; si[i] = pi[i]; }
  __syncthreads();

  for (int q = 0; q < NQ; ++q) {
    const int pos = 9 - q, bit = 1 << pos;
    float ax = 0.f, ay = 0.f, az = 0.f;
    for (int p = t; p < DIM / 2; p += 256) {
      int i0 = ((p >> pos) << (pos + 1)) | (p & (bit - 1));
      int i1 = i0 | bit;
      float ar = sr[i0], ai = si[i0], br = sr[i1], bi = si[i1];
      ax += ar * br + ai * bi;                  // Re(conj(a)*b)
      ay += ar * bi - ai * br;                  // Im(conj(a)*b)
      az += ar * ar + ai * ai - br * br - bi * bi;
    }
    rx[t] = ax; ry[t] = ay; rz[t] = az;
    __syncthreads();
    for (int s = 128; s > 0; s >>= 1) {
      if (t < s) { rx[t] += rx[t + s]; ry[t] += ry[t + s]; rz[t] += rz[t + s]; }
      __syncthreads();
    }
    if (t == 0) {
      out[(size_t)row * 30 + q]       = 2.0f * rx[0];
      out[(size_t)row * 30 + 10 + q]  = 2.0f * ry[0];
      out[(size_t)row * 30 + 20 + q]  = rz[0];
    }
    __syncthreads();
  }
}

// ---------------------------------------------------------------------------
extern "C" void kernel_launch(void* const* d_in, const int* in_sizes, int n_in,
                              void* d_out, int out_size, void* d_ws, size_t ws_size,
                              hipStream_t stream) {
  const float* x = (const float*)d_in[0];
  const float* w = (const float*)d_in[1];
  float* out = (float*)d_out;

  const int B = in_sizes[0] / XLEN;             // 8192
  const size_t nS = (size_t)B * DIM;
  const size_t nF = (size_t)DIM * DIM;

  char* ws = (char*)d_ws;
  _Float16* Sr = (_Float16*)ws;                 // B*1024 f16
  _Float16* Si = Sr + nS;
  _Float16* Fr = Si + nS;                       // 1024*1024 f16
  _Float16* Fi = Fr + nF;
  float* Or = (float*)(Fi + nF);                // B*1024 f32
  float* Oi = Or + nS;

  k_prep<<<B, 256, 0, stream>>>(x, w, Sr, Si);
  k_qft<<<(int)(nF / 256), 256, 0, stream>>>(Fr, Fi);
  k_gemm<<<dim3(B / 128, DIM / 64), 256, 0, stream>>>(Sr, Si, Fr, Fi, Or, Oi);
  k_expect<<<B, 256, 0, stream>>>(Or, Oi, out);
}
